// PanguAttention_46634754900592
// MI455X (gfx1250) — compile-verified, matrix operand order checked by reading, not run on hardware
//
#include <hip/hip_runtime.h>
#include <hip/hip_bf16.h>

typedef __attribute__((ext_vector_type(16))) __bf16 v16bf;
typedef __attribute__((ext_vector_type(8)))  __bf16 v8bf;
typedef __attribute__((ext_vector_type(8)))  float  v8f;
typedef __attribute__((ext_vector_type(4)))  float  v4f;

static constexpr int Bn   = 2;
static constexpr int Sn   = 2048;
static constexpr int HIDn = 4096;
static constexpr int Hn   = 32;
static constexpr int KVHn = 8;
static constexpr int Dn   = 128;
static constexpr float SCALEc = 0.08838834764831845f;   // D^-0.5
static constexpr float THETAc = 10000.0f;

__device__ inline v8f vzero8() {
  v8f z;
#pragma unroll
  for (int i = 0; i < 8; ++i) z[i] = 0.0f;
  return z;
}

// ---------------------------------------------------------------------------
// fp32 -> bf16 conversion, 8 elements / thread
// ---------------------------------------------------------------------------
__global__ __launch_bounds__(256) void cvt_f32_bf16(const float* __restrict__ src,
                                                    __bf16* __restrict__ dst, int n8) {
  int i = blockIdx.x * blockDim.x + threadIdx.x;
  if (i >= n8) return;
  const v4f* s = (const v4f*)src + (size_t)i * 2;
  v4f a = s[0], b = s[1];
  v8bf o;
#pragma unroll
  for (int j = 0; j < 4; ++j) { o[j] = (__bf16)a[j]; o[4 + j] = (__bf16)b[j]; }
  ((v8bf*)dst)[i] = o;
}

// ---------------------------------------------------------------------------
// fp32 [K x N] -> bf16 transposed [N x K], 64x64 tile via LDS.
// One-time pass over the weights so GEMM B staging is fully vectorized.
// ---------------------------------------------------------------------------
__global__ __launch_bounds__(256) void cvt_transpose_f32_bf16(const float* __restrict__ src,
                                                              __bf16* __restrict__ dst,
                                                              int K, int N) {
  __shared__ __bf16 T[64 * 72];   // [k][n], pad to 72 halves
  const int k0 = blockIdx.y * 64, n0 = blockIdx.x * 64;
  const int t = threadIdx.x;
  {
    const int kr = t >> 2, cg = (t & 3) * 16;
    const float* s = &src[(size_t)(k0 + kr) * N + n0 + cg];
#pragma unroll
    for (int j = 0; j < 4; ++j) {
      v4f a = *(const v4f*)(s + j * 4);
#pragma unroll
      for (int e = 0; e < 4; ++e) T[kr * 72 + cg + j * 4 + e] = (__bf16)a[e];
    }
  }
  __syncthreads();
  {
    const int nr = t >> 2, kg = (t & 3) * 16;
    v8bf o0, o1;
#pragma unroll
    for (int e = 0; e < 8; ++e) {
      o0[e] = T[(kg + e) * 72 + nr];
      o1[e] = T[(kg + 8 + e) * 72 + nr];
    }
    *(v8bf*)&dst[(size_t)(n0 + nr) * K + k0 + kg]     = o0;
    *(v8bf*)&dst[(size_t)(n0 + nr) * K + k0 + kg + 8] = o1;
  }
}

// ---------------------------------------------------------------------------
// bf16 GEMM: C[M,N] = A[M,K] x B[K,N]; A row-major, B given PRE-TRANSPOSED
// as BT[N,K]. Block tile 128(M) x 128(N), 256 threads = 8 waves (4x2), wave
// tile 32x64 (2 A-frags x 4 B-frags -> 8 WMMA from 12 b128 loads / k-step).
// K staged in LDS in steps of 32 -> v_wmma_f32_16x16x32_bf16.
// OUTMODE: 0 = bf16 row-major, 1 = bf16 transposed C[n*M+m], 2 = f32 row-major
// ---------------------------------------------------------------------------
template <int OUTMODE>
__global__ __launch_bounds__(256) void gemm_bf16(const __bf16* __restrict__ A,
                                                 const __bf16* __restrict__ BT,
                                                 void* __restrict__ Cout,
                                                 int M, int N, int K) {
  __shared__ __bf16 As[128 * 40];   // row stride 40 halves (80 B, keeps 16B align)
  __shared__ __bf16 Bs[128 * 40];   // Bs[n][k]

  const int t    = threadIdx.x;
  const int m0b  = blockIdx.y * 128;
  const int n0b  = blockIdx.x * 128;
  const int wave = t >> 5;
  const int lane = t & 31;
  const int half = lane >> 4;
  const int ln   = lane & 15;
  const int wm   = (wave >> 1) * 32;   // 0,32,64,96
  const int wn   = (wave & 1) * 64;    // 0,64

  v8f c[2][4];
#pragma unroll
  for (int i = 0; i < 2; ++i)
#pragma unroll
    for (int j = 0; j < 4; ++j) c[i][j] = vzero8();

  const int ldr = t >> 1, ldg = (t & 1) * 16;   // tile loaders: 128 rows x 32 k

  for (int k0 = 0; k0 < K; k0 += 32) {
    // ---- stage A 128x32 and B 128x32 (two v8bf each per thread) ----
    {
      const __bf16* arow = &A[(size_t)(m0b + ldr) * K + k0 + ldg];
      *(v8bf*)&As[ldr * 40 + ldg]     = *(const v8bf*)(arow);
      *(v8bf*)&As[ldr * 40 + ldg + 8] = *(const v8bf*)(arow + 8);
      const __bf16* brow = &BT[(size_t)(n0b + ldr) * K + k0 + ldg];
      *(v8bf*)&Bs[ldr * 40 + ldg]     = *(const v8bf*)(brow);
      *(v8bf*)&Bs[ldr * 40 + ldg + 8] = *(const v8bf*)(brow + 8);
    }
    __syncthreads();

    if (k0 + 32 < K) {  // prefetch next tiles into L2 (global_prefetch_b8)
      __builtin_prefetch(&A[(size_t)(m0b + ldr) * K + k0 + 32 + ldg], 0, 1);
      __builtin_prefetch(&BT[(size_t)(n0b + ldr) * K + k0 + 32 + ldg], 0, 1);
    }

    // ---- fragments (ISA 16-bit A/B layout: k = i + 8*(i>=8) + 8*half) ----
    const int ko = half * 8;
    v16bf a[2], b[4];
#pragma unroll
    for (int i = 0; i < 2; ++i) {
      const __bf16* ar = &As[(wm + i * 16 + ln) * 40];
      ((v8bf*)&a[i])[0] = *(const v8bf*)(ar + ko);
      ((v8bf*)&a[i])[1] = *(const v8bf*)(ar + 16 + ko);
    }
#pragma unroll
    for (int j = 0; j < 4; ++j) {
      const __bf16* br = &Bs[(wn + j * 16 + ln) * 40];
      ((v8bf*)&b[j])[0] = *(const v8bf*)(br + ko);
      ((v8bf*)&b[j])[1] = *(const v8bf*)(br + 16 + ko);
    }

#pragma unroll
    for (int i = 0; i < 2; ++i)
#pragma unroll
      for (int j = 0; j < 4; ++j)
        c[i][j] = __builtin_amdgcn_wmma_f32_16x16x32_bf16(false, a[i], false, b[j],
                                                          (short)0, c[i][j], false, false);
    __syncthreads();
  }

  // ---- store: C layout -> lane holds column n, reg r holds row r + 8*half ----
  const int mA = m0b + wm, nA = n0b + wn;
#pragma unroll
  for (int i = 0; i < 2; ++i) {
#pragma unroll
    for (int j = 0; j < 4; ++j) {
      const int nc = nA + j * 16 + ln;
      if constexpr (OUTMODE == 1) {
        // transposed store: 8 consecutive rows per lane -> one v8bf per tile
        __bf16* C = (__bf16*)Cout;
        v8bf tv;
#pragma unroll
        for (int r = 0; r < 8; ++r) tv[r] = (__bf16)c[i][j][r];
        *(v8bf*)&C[(size_t)nc * M + mA + i * 16 + 8 * half] = tv;
      } else if constexpr (OUTMODE == 2) {
        float* C = (float*)Cout;
#pragma unroll
        for (int r = 0; r < 8; ++r)
          C[(size_t)(mA + i * 16 + r + 8 * half) * N + nc] = c[i][j][r];
      } else {
        __bf16* C = (__bf16*)Cout;
#pragma unroll
        for (int r = 0; r < 8; ++r)
          C[(size_t)(mA + i * 16 + r + 8 * half) * N + nc] = (__bf16)c[i][j][r];
      }
    }
  }
}

// ---------------------------------------------------------------------------
// RoPE in place on bf16 [B,S,nh,D]; one thread per (b,s,h,d<64) pair.
// ---------------------------------------------------------------------------
__global__ __launch_bounds__(256) void rope_bf16(__bf16* __restrict__ x,
                                                 const int* __restrict__ positions,
                                                 int nh, int total) {
  int idx = blockIdx.x * blockDim.x + threadIdx.x;
  if (idx >= total) return;
  const int dh = idx & 63;
  int r = idx >> 6;
  const int h = r % nh; r /= nh;
  const int s = r % Sn;
  const int b = r / Sn;
  const float pos  = (float)positions[b * Sn + s];
  const float freq = __powf(THETAc, -(float)dh * (1.0f / 64.0f));
  const float ang  = pos * freq;
  const float c = __cosf(ang), sn = __sinf(ang);
  const size_t base = (((size_t)(b * Sn + s)) * nh + h) * Dn;
  const float x1 = (float)x[base + dh];
  const float x2 = (float)x[base + 64 + dh];
  x[base + dh]      = (__bf16)(x1 * c - x2 * sn);
  x[base + 64 + dh] = (__bf16)(x2 * c + x1 * sn);
}

// ---------------------------------------------------------------------------
// Flash attention (causal, GQA 4:1). 128 threads = 4 waves; block owns 64 q
// rows of one (b,h); each wave owns 16 q rows. Streams KV in 32-row tiles.
// K is [B,S,KVH,D] row-major; V comes PRE-TRANSPOSED as Vt[kvh*D+d][b*S+s]
// so both LDS stagings are pure v8bf copies.
// Computes S^T = K*Q^T so softmax is column(=query)-local per lane, and the
// C-layout of S^T *is* the B-operand layout of O^T = V^T * P^T (no shuffles).
// ---------------------------------------------------------------------------
__global__ __launch_bounds__(128) void flash_attn(const __bf16* __restrict__ Q,
                                                  const __bf16* __restrict__ Kc,
                                                  const __bf16* __restrict__ Vt_g,
                                                  __bf16* __restrict__ O) {
  __shared__ __bf16 Ks[32 * 136];   // K tile row-major, row stride 136 halves
  __shared__ __bf16 Vt[128 * 40];   // V tile transposed: Vt[d][kv]

  const int t    = threadIdx.x;
  const int wave = t >> 5;
  const int lane = t & 31;
  const int half = lane >> 4;
  const int ln   = lane & 15;

  const int bh  = blockIdx.y;
  const int b   = bh / Hn;
  const int h   = bh % Hn;
  const int kvh = h / (Hn / KVHn);
  const int qb  = blockIdx.x * 64;
  const int qw  = qb + wave * 16;
  const int q   = qw + ln;               // this lane's query row (both halves)
  const int BS  = Bn * Sn;

  // Q fragments = B operand of S^T (lane = column q; k over d)
  v16bf bq[4];
  {
    const __bf16* qrow = &Q[(((size_t)(b * Sn + q)) * Hn + h) * Dn];
#pragma unroll
    for (int dc = 0; dc < 4; ++dc) {
      ((v8bf*)&bq[dc])[0] = *(const v8bf*)(qrow + dc * 32 + half * 8);
      ((v8bf*)&bq[dc])[1] = *(const v8bf*)(qrow + dc * 32 + 16 + half * 8);
    }
  }

  float m_run = -1e30f, l_run = 0.0f;
  v8f acc[8];
#pragma unroll
  for (int td = 0; td < 8; ++td) acc[td] = vzero8();

  const int ldr = t >> 2, ldc = (t & 3) * 32;    // K tile loader (32 rows x 128)
  const __bf16* vbase = &Vt_g[((size_t)(kvh * Dn + t)) * BS + b * Sn];  // V: row d = t
  const int nblk = (qb + 64) / 32;

  for (int kb = 0; kb < nblk; ++kb) {
    const int kv0 = kb * 32;
    // ---- stage K 32x128 row-major and V 128x32 (already transposed) ----
    {
      const __bf16* ksrc = &Kc[(((size_t)(b * Sn + kv0 + ldr)) * KVHn + kvh) * Dn + ldc];
#pragma unroll
      for (int j = 0; j < 4; ++j)
        *(v8bf*)&Ks[ldr * 136 + ldc + j * 8] = *(const v8bf*)(ksrc + j * 8);
      const __bf16* vsrc = vbase + kv0;
#pragma unroll
      for (int j = 0; j < 4; ++j)
        *(v8bf*)&Vt[t * 40 + j * 8] = *(const v8bf*)(vsrc + j * 8);
    }
    __syncthreads();

    // ---- S^T tiles: (kv 0-15) and (kv 16-31) x (16 q), accumulate over d ----
    v8f st[2];
#pragma unroll
    for (int tt = 0; tt < 2; ++tt) {
      st[tt] = vzero8();
      const __bf16* krow = &Ks[(tt * 16 + ln) * 136];
#pragma unroll
      for (int dc = 0; dc < 4; ++dc) {
        v16bf ak;
        ((v8bf*)&ak)[0] = *(const v8bf*)(krow + dc * 32 + half * 8);
        ((v8bf*)&ak)[1] = *(const v8bf*)(krow + dc * 32 + 16 + half * 8);
        st[tt] = __builtin_amdgcn_wmma_f32_16x16x32_bf16(false, ak, false, bq[dc],
                                                         (short)0, st[tt], false, false);
      }
    }

    // ---- scale + causal mask (lane column = q; reg r = kv row) ----
    float s[16];
#pragma unroll
    for (int tt = 0; tt < 2; ++tt)
#pragma unroll
      for (int r = 0; r < 8; ++r) {
        const int kv = kv0 + tt * 16 + r + 8 * half;
        const float v = st[tt][r] * SCALEc;
        s[tt * 8 + r] = (kv <= q) ? v : -1e30f;
      }

    // ---- online softmax (per query column) ----
    float bm = s[0];
#pragma unroll
    for (int i = 1; i < 16; ++i) bm = fmaxf(bm, s[i]);
    bm = fmaxf(bm, __shfl_xor(bm, 16, 32));        // combine the two lane halves
    const float m_new = fmaxf(m_run, bm);
    const float corr  = __expf(m_run - m_new);
    float p[16];
    float bs = 0.0f;
#pragma unroll
    for (int i = 0; i < 16; ++i) { p[i] = __expf(s[i] - m_new); bs += p[i]; }
    bs += __shfl_xor(bs, 16, 32);
    l_run = l_run * corr + bs;
    m_run = m_new;
#pragma unroll
    for (int td = 0; td < 8; ++td)
#pragma unroll
      for (int e = 0; e < 8; ++e) acc[td][e] *= corr;

    // ---- P^T fragment: C-layout of S^T == B-operand layout (identity map) ----
    v16bf bp;
#pragma unroll
    for (int i = 0; i < 16; ++i) bp[i] = (__bf16)p[i];

    // ---- O^T += V^T * P^T : 8 d-tiles of 16x16 ----
#pragma unroll
    for (int td = 0; td < 8; ++td) {
      const __bf16* vrow = &Vt[(td * 16 + ln) * 40];
      v16bf av;
      ((v8bf*)&av)[0] = *(const v8bf*)(vrow + half * 8);
      ((v8bf*)&av)[1] = *(const v8bf*)(vrow + 16 + half * 8);
      acc[td] = __builtin_amdgcn_wmma_f32_16x16x32_bf16(false, av, false, bp,
                                                        (short)0, acc[td], false, false);
    }
    __syncthreads();
  }

  // ---- normalize + store: lane column = q, reg r = d row ----
  const float inv = 1.0f / l_run;
  __bf16* orow = &O[(((size_t)(b * Sn + q)) * Hn + h) * Dn];
#pragma unroll
  for (int td = 0; td < 8; ++td)
#pragma unroll
    for (int r = 0; r < 8; ++r) {
      const int d = td * 16 + r + 8 * half;
      orow[d] = (__bf16)(acc[td][r] * inv);
    }
}

// ---------------------------------------------------------------------------
// launcher
// ---------------------------------------------------------------------------
extern "C" void kernel_launch(void* const* d_in, const int* in_sizes, int n_in,
                              void* d_out, int out_size, void* d_ws, size_t ws_size,
                              hipStream_t stream) {
  (void)in_sizes; (void)n_in; (void)out_size; (void)ws_size;
  const int*   positions = (const int*)d_in[0];
  const float* hidden    = (const float*)d_in[1];
  const float* wq        = (const float*)d_in[2];
  const float* wk        = (const float*)d_in[3];
  const float* wv        = (const float*)d_in[4];
  const float* wo        = (const float*)d_in[5];

  const int BS = Bn * Sn;            // 4096
  const int NQ = Hn * Dn;            // 4096
  const int NK = KVHn * Dn;          // 1024

  char* ws = (char*)d_ws;
  size_t off = 0;
  auto alloc = [&](size_t elems) -> __bf16* {
    __bf16* p = (__bf16*)(ws + off);
    off += elems * 2;
    off = (off + 255) & ~(size_t)255;
    return p;
  };
  __bf16* Xb   = alloc((size_t)BS * HIDn);     // hidden, row-major
  __bf16* WqT  = alloc((size_t)NQ * HIDn);     // wq^T  [N x K]
  __bf16* WkT  = alloc((size_t)NK * HIDn);     // wk^T
  __bf16* WvT  = alloc((size_t)NK * HIDn);     // wv^T
  __bf16* WoT  = alloc((size_t)HIDn * NQ);     // wo^T
  __bf16* Qb   = alloc((size_t)BS * NQ);       // q, row-major [BS, H*D]
  __bf16* Kb   = alloc((size_t)BS * NK);       // k, row-major [BS, KVH*D]
  __bf16* Vt_g = alloc((size_t)NK * BS);       // v transposed [KVH*D, BS]
  __bf16* Ab   = alloc((size_t)BS * NQ);       // attention out, row-major

  // fp32 -> bf16 (hidden stays row-major; weights transposed once)
  {
    int n8 = (BS * HIDn) / 8;
    cvt_f32_bf16<<<(n8 + 255) / 256, 256, 0, stream>>>(hidden, Xb, n8);
  }
  cvt_transpose_f32_bf16<<<dim3(NQ / 64, HIDn / 64), 256, 0, stream>>>(wq, WqT, HIDn, NQ);
  cvt_transpose_f32_bf16<<<dim3(NK / 64, HIDn / 64), 256, 0, stream>>>(wk, WkT, HIDn, NK);
  cvt_transpose_f32_bf16<<<dim3(NK / 64, HIDn / 64), 256, 0, stream>>>(wv, WvT, HIDn, NK);
  cvt_transpose_f32_bf16<<<dim3(HIDn / 64, NQ / 64), 256, 0, stream>>>(wo, WoT, NQ, HIDn);

  // QKV projections (V written transposed for flash staging)
  gemm_bf16<0><<<dim3(NQ / 128, BS / 128), 256, 0, stream>>>(Xb, WqT, Qb, BS, NQ, HIDn);
  gemm_bf16<0><<<dim3(NK / 128, BS / 128), 256, 0, stream>>>(Xb, WkT, Kb, BS, NK, HIDn);
  gemm_bf16<1><<<dim3(NK / 128, BS / 128), 256, 0, stream>>>(Xb, WvT, Vt_g, BS, NK, HIDn);

  // RoPE on Q and K (V needs none)
  {
    int tq = BS * Hn * 64;
    rope_bf16<<<(tq + 255) / 256, 256, 0, stream>>>(Qb, positions, Hn, tq);
    int tk = BS * KVHn * 64;
    rope_bf16<<<(tk + 255) / 256, 256, 0, stream>>>(Kb, positions, KVHn, tk);
  }

  // causal flash attention
  flash_attn<<<dim3(Sn / 64, Bn * Hn), 128, 0, stream>>>(Qb, Kb, Vt_g, Ab);

  // output projection -> f32 out
  gemm_bf16<2><<<dim3(HIDn / 128, BS / 128), 256, 0, stream>>>(Ab, WoT, d_out, BS, HIDn, NQ);
}
